// MultiHeadAttention_21741124452725
// MI455X (gfx1250) — matrix operand order repaired and checked by reading, not compiled, not run on hardware
//
#include <hip/hip_runtime.h>

// ---------------------------------------------------------------------------
// MultiHeadAttention for MI455X (gfx1250, wave32, WMMA 16x16x32 bf16,
// async global->LDS staging, double-buffered GEMM pipeline)
// B=2, N=2048, D_IN=D_OUT=1024, H=16, HEAD_DIM=64
// ---------------------------------------------------------------------------

#define BATCH   2
#define SEQ     2048
#define DMODEL  1024
#define NHEAD   16
#define HDIM    64
#define BN_TOK  (BATCH * SEQ)      // 4096 tokens total

typedef __attribute__((ext_vector_type(16))) __bf16        v16bf;
typedef __attribute__((ext_vector_type(8)))  float         v8f;
typedef __attribute__((ext_vector_type(4)))  unsigned int  u32x4;

// ---- helpers --------------------------------------------------------------

__device__ __forceinline__ unsigned short f2bf(float f) {
  // round-to-nearest-even fp32 -> bf16
  unsigned int u = __builtin_bit_cast(unsigned int, f);
  unsigned int r = ((u >> 16) & 1u) + 0x7FFFu;
  return (unsigned short)((u + r) >> 16);
}

struct Frag2 { u32x4 a, b; };

__device__ __forceinline__ v16bf mkfrag(u32x4 a, u32x4 b) {
  Frag2 t{a, b};
  return __builtin_bit_cast(v16bf, t);
}

// A-fragment (16x32 bf16): two contiguous 8-elem (16B) runs
__device__ __forceinline__ v16bf frag_runs(const unsigned short* p0,
                                           const unsigned short* p1) {
  return mkfrag(*(const u32x4*)p0, *(const u32x4*)p1);
}
// B-fragment (32x16 bf16): one contiguous 16-elem (32B) run
__device__ __forceinline__ v16bf frag32(const unsigned short* p) {
  return mkfrag(((const u32x4*)p)[0], ((const u32x4*)p)[1]);
}

__device__ __forceinline__ v8f wmma_bf16(v16bf a, v16bf b, v8f c) {
  // (neg_a, A, neg_b, B, c_mod, C, reuse_a, reuse_b)
  return __builtin_amdgcn_wmma_f32_16x16x32_bf16(false, a, false, b,
                                                 (short)0, c, false, false);
}

// CDNA5 async global->LDS copy, 16B per lane, tracked on ASYNCcnt.
// LDS operand: flat pointer to __shared__ truncates to the LDS byte offset
// (see ISA 10.2, LDS aperture: low 32 address bits are used).
__device__ __forceinline__ void async_b128(const unsigned short* gsrc,
                                           unsigned short* ldst) {
  unsigned loff = (unsigned)(uintptr_t)ldst;
  asm volatile("global_load_async_to_lds_b128 %0, %1, off"
               :: "v"(loff), "v"(gsrc) : "memory");
}
__device__ __forceinline__ void wait_async0() {
  asm volatile("s_wait_asynccnt 0x0" ::: "memory");
}

// ---- kernel 0a: fp32 -> bf16 conversion (x) --------------------------------

__global__ void cvt_bf16(const float* __restrict__ src,
                         unsigned short* __restrict__ dst, int n) {
  int i = blockIdx.x * blockDim.x + threadIdx.x;
  int stride = gridDim.x * blockDim.x;
  for (; i < n; i += stride) dst[i] = f2bf(src[i]);
}

// ---- kernel 0b: fp32 -> bf16 with transpose (weights) ----------------------
// W[k][n] fp32  ->  Wt[n][k] bf16, so GEMM B-tiles stage with contiguous b128
// copies and B-fragments read contiguous [n][k] runs.

__global__ void cvt_w_t(const float* __restrict__ src,
                        unsigned short* __restrict__ dst) {
  int i = blockIdx.x * blockDim.x + threadIdx.x;
  int stride = gridDim.x * blockDim.x;
  for (; i < DMODEL * DMODEL; i += stride) {
    const int k = i >> 10, n = i & (DMODEL - 1);
    dst[(size_t)n * DMODEL + k] = f2bf(src[i]);
  }
}

// ---- kernel 1: fused QKV projection GEMM -----------------------------------
// grid = (BN_TOK/128, DMODEL/128, 3[q,k,v]); block = 256 (8 waves)
// workgroup tile 128x128; wave tile 32x64 (2x4 wmma accumulators)
// Async-staged, double-buffered k-pipeline (tile k+1 streams to LDS while
// WMMAs consume tile k). Q is pre-scaled by 1/sqrt(HDIM) (exact in bf16).
// Q,K stored row-major [token][DMODEL] bf16; V stored transposed per head
// Vt[b][h][d][n] bf16 so the PV B-fragments are contiguous.

__device__ __forceinline__ void stage_tiles(
    const unsigned short* __restrict__ xb,
    const unsigned short* __restrict__ Wt,
    unsigned short* lA, unsigned short* lB,
    int tileM, int tileN, int k0, int tid)
{
  const int m = tid >> 1, half = tid & 1;
  // A tile 128x32 (row-major [m][k]): 32B per thread = 2 async b128
  {
    const unsigned short* g = xb + (size_t)(tileM + m) * DMODEL + k0 + half * 16;
    unsigned short* s = lA + m * 32 + half * 16;
    async_b128(g,     s);
    async_b128(g + 8, s + 8);
  }
  // B tile 128x32 ([n][k] from transposed weights): same pattern
  {
    const unsigned short* g = Wt + (size_t)(tileN + m) * DMODEL + k0 + half * 16;
    unsigned short* s = lB + m * 32 + half * 16;
    async_b128(g,     s);
    async_b128(g + 8, s + 8);
  }
}

__global__ __launch_bounds__(256) void gemm_qkv(
    const unsigned short* __restrict__ xb,   // [4096][1024] bf16
    const unsigned short* __restrict__ wqt,  // [1024][1024] bf16, transposed
    const unsigned short* __restrict__ wkt,
    const unsigned short* __restrict__ wvt,
    unsigned short* __restrict__ Qb,         // [4096][1024] (pre-scaled)
    unsigned short* __restrict__ Kb,         // [4096][1024]
    unsigned short* __restrict__ Vt)         // [B][H][64][2048]
{
  const int which = blockIdx.z;
  const unsigned short* __restrict__ Wt =
      (which == 0) ? wqt : (which == 1) ? wkt : wvt;

  const int tileM = blockIdx.x * 128;
  const int tileN = blockIdx.y * 128;

  __shared__ __align__(16) unsigned short lA[2][128 * 32];  // [m][k]
  __shared__ __align__(16) unsigned short lB[2][128 * 32];  // [n][k]

  const int tid  = threadIdx.x;
  const int lane = tid & 31;
  const int wave = tid >> 5;
  const int wm   = (wave & 3) * 32;   // wave M offset (4 waves along M)
  const int wn   = (wave >> 2) * 64;  // wave N offset (2 waves along N)
  const int lm   = lane & 15;
  const int hi   = lane >> 4;

  v8f acc[2][4];
#pragma unroll
  for (int i = 0; i < 2; ++i)
#pragma unroll
    for (int j = 0; j < 4; ++j) acc[i][j] = v8f{0.f,0.f,0.f,0.f,0.f,0.f,0.f,0.f};

  stage_tiles(xb, Wt, lA[0], lB[0], tileM, tileN, 0, tid);

  const int KT = DMODEL / 32;  // 32 k-steps
  for (int kt = 0; kt < KT; ++kt) {
    const int p = kt & 1;
    wait_async0();      // drain async writes into buffer p
    __syncthreads();    // everyone's tile p ready; tile p^1 free to overwrite
    if (kt + 1 < KT)
      stage_tiles(xb, Wt, lA[p ^ 1], lB[p ^ 1], tileM, tileN, (kt + 1) * 32, tid);

    // A fragments: rows wm + i*16 + lm; 16 contiguous K at hi*16
    v16bf af[2];
#pragma unroll
    for (int i = 0; i < 2; ++i) {
      const unsigned short* base = lA[p] + (wm + i * 16 + lm) * 32;
      af[i] = frag_runs(base + hi * 16, base + hi * 16 + 8);
    }
    // B fragments: cols wn + j*16 + lm; 16 contiguous K at hi*16
    v16bf bfr[4];
#pragma unroll
    for (int j = 0; j < 4; ++j)
      bfr[j] = frag32(lB[p] + (wn + j * 16 + lm) * 32 + hi * 16);

#pragma unroll
    for (int i = 0; i < 2; ++i)
#pragma unroll
      for (int j = 0; j < 4; ++j) acc[i][j] = wmma_bf16(af[i], bfr[j], acc[i][j]);
  }

  // epilogue
  if (which < 2) {
    unsigned short* __restrict__ out = (which == 0) ? Qb : Kb;
    const float sc = (which == 0) ? 0.125f : 1.0f;   // fold 1/sqrt(64) into Q
#pragma unroll
    for (int i = 0; i < 2; ++i)
#pragma unroll
      for (int j = 0; j < 4; ++j) {
        const int col = tileN + wn + j * 16 + lm;
        const int rowb = tileM + wm + i * 16 + hi * 8;
#pragma unroll
        for (int r = 0; r < 8; ++r)
          out[(size_t)(rowb + r) * DMODEL + col] = f2bf(acc[i][j][r] * sc);
      }
  } else {
    // V: store transposed per head. lane has fixed dim, 8 consecutive tokens
#pragma unroll
    for (int i = 0; i < 2; ++i)
#pragma unroll
      for (int j = 0; j < 4; ++j) {
        const int token = tileM + wm + i * 16 + hi * 8;  // +r contiguous
        const int bb = token >> 11, n = token & (SEQ - 1);
        const int dim = tileN + wn + j * 16 + lm;
        const int hh = dim >> 6, dd = dim & 63;
        unsigned short* pp = Vt + ((size_t)((bb * NHEAD + hh) * HDIM + dd) << 11) + n;
        unsigned int w0 = f2bf(acc[i][j][0]) | ((unsigned int)f2bf(acc[i][j][1]) << 16);
        unsigned int w1 = f2bf(acc[i][j][2]) | ((unsigned int)f2bf(acc[i][j][3]) << 16);
        unsigned int w2 = f2bf(acc[i][j][4]) | ((unsigned int)f2bf(acc[i][j][5]) << 16);
        unsigned int w3 = f2bf(acc[i][j][6]) | ((unsigned int)f2bf(acc[i][j][7]) << 16);
        u32x4 pv{w0, w1, w2, w3};
        *(u32x4*)pp = pv;
      }
  }
}

// ---- kernel 2: causal flash attention --------------------------------------
// grid = (SEQ/128, B*H); block = 256 (8 waves). Wave owns 16 query rows,
// streams 32-key blocks up to the causal frontier. fp32 online softmax.
// Q arrives pre-scaled by 1/sqrt(HDIM). No workgroup barriers in the
// (divergent-length) key loop: LDS P-slab is per-wave, DS ops are in-order.

__global__ __launch_bounds__(256) void attn(
    const unsigned short* __restrict__ Qb,   // [4096][1024] (pre-scaled)
    const unsigned short* __restrict__ Kb,   // [4096][1024]
    const unsigned short* __restrict__ Vt,   // [B][H][64][2048]
    float* __restrict__ out)                 // [B][N][1024] fp32
{
  const int bh = blockIdx.y;                 // 0..31
  const int b = bh >> 4, h = bh & 15;
  const int tid  = threadIdx.x;
  const int lane = tid & 31;
  const int wave = tid >> 5;
  const int lm   = lane & 15;
  const int hi   = lane >> 4;
  const int qbase = blockIdx.x * 128 + wave * 16;

  // per-wave LDS slab for P (C-layout -> A-layout transpose), 16x32 bf16
  __shared__ __align__(16) unsigned short pshm[8 * 16 * 32];
  unsigned short* pbuf = pshm + wave * 512;

  // Q fragments (queries qbase..qbase+15, head h), two 32-wide d-steps
  const unsigned short* qrow =
      Qb + (size_t)(b * SEQ + qbase + lm) * DMODEL + h * HDIM;
  const v16bf qf0 = frag_runs(qrow + hi * 16,      qrow + hi * 16 + 8);   // d 0..31
  const v16bf qf1 = frag_runs(qrow + 32 + hi * 16, qrow + 40 + hi * 16);  // d 32..63

  v8f o[4];
#pragma unroll
  for (int j = 0; j < 4; ++j) o[j] = v8f{0.f,0.f,0.f,0.f,0.f,0.f,0.f,0.f};
  float mrow[8], lrow[8];
#pragma unroll
  for (int r = 0; r < 8; ++r) { mrow[r] = -1e30f; lrow[r] = 0.f; }

  const int nkb = (qbase + 16 + 31) >> 5;

  for (int kb0 = 0; kb0 < nkb; ++kb0) {
    const int kb = kb0 * 32;

    // K B-fragments: lane = key token, elems = 16 contiguous head dims
    const unsigned short* kr0 = Kb + (size_t)(b * SEQ + kb + lm) * DMODEL + h * HDIM;
    const unsigned short* kr1 = Kb + (size_t)(b * SEQ + kb + 16 + lm) * DMODEL + h * HDIM;
    v16bf k00 = frag32(kr0 + hi * 16);        // keys kb..+15, d 0..31
    v16bf k01 = frag32(kr0 + 32 + hi * 16);   // keys kb..+15, d 32..63
    v16bf k10 = frag32(kr1 + hi * 16);        // keys kb+16..+31, d 0..31
    v16bf k11 = frag32(kr1 + 32 + hi * 16);

    if (kb0 + 1 < nkb) {
      __builtin_prefetch(kr0 + (size_t)32 * DMODEL, 0, 1);
      __builtin_prefetch(Vt + ((size_t)(bh * HDIM + lm) << 11) + kb + 32, 0, 1);
    }

    v8f s0 = v8f{0.f,0.f,0.f,0.f,0.f,0.f,0.f,0.f};
    v8f s1 = v8f{0.f,0.f,0.f,0.f,0.f,0.f,0.f,0.f};
    s0 = wmma_bf16(qf0, k00, s0);
    s0 = wmma_bf16(qf1, k01, s0);
    s1 = wmma_bf16(qf0, k10, s1);
    s1 = wmma_bf16(qf1, k11, s1);

    const int col0 = kb + lm, col1 = col0 + 16;
#pragma unroll
    for (int r = 0; r < 8; ++r) {
      const int q = qbase + hi * 8 + r;
      float a = (col0 <= q) ? s0[r] : -1e30f;   // Q pre-scaled: no mult here
      float c = (col1 <= q) ? s1[r] : -1e30f;
      // row max across the 16 column lanes of this half
      float t = fmaxf(a, c);
      t = fmaxf(t, __shfl_xor(t, 1));
      t = fmaxf(t, __shfl_xor(t, 2));
      t = fmaxf(t, __shfl_xor(t, 4));
      t = fmaxf(t, __shfl_xor(t, 8));
      const float mn = fmaxf(mrow[r], t);
      const float f  = __expf(mrow[r] - mn);
      const float p0 = __expf(a - mn);
      const float p1 = __expf(c - mn);
      float rs = p0 + p1;
      rs += __shfl_xor(rs, 1);
      rs += __shfl_xor(rs, 2);
      rs += __shfl_xor(rs, 4);
      rs += __shfl_xor(rs, 8);
      lrow[r] = lrow[r] * f + rs;
      mrow[r] = mn;
      o[0][r] *= f; o[1][r] *= f; o[2][r] *= f; o[3][r] *= f;
      // spill P (C-layout) to per-wave LDS, row-major 16x32
      pbuf[(hi * 8 + r) * 32 + lm]      = f2bf(p0);
      pbuf[(hi * 8 + r) * 32 + lm + 16] = f2bf(p1);
    }

    // reload P as an A-fragment (wave-internal LDS round trip; DS in-order)
    const unsigned short* pr = pbuf + lm * 32;
    v16bf pf = frag_runs(pr + hi * 16, pr + hi * 16 + 8);

    // PV: B-fragments from transposed V, 16 contiguous key tokens per lane
#pragma unroll
    for (int j = 0; j < 4; ++j) {
      const unsigned short* vr =
          Vt + ((size_t)(bh * HDIM + j * 16 + lm) << 11) + kb + hi * 16;
      o[j] = wmma_bf16(pf, frag32(vr), o[j]);
    }
  }

  // epilogue: normalize and store fp32
#pragma unroll
  for (int r = 0; r < 8; ++r) {
    const float inv = 1.0f / lrow[r];
    const int row = b * SEQ + qbase + hi * 8 + r;
    float* op = out + (size_t)row * DMODEL + h * HDIM + lm;
    op[0]  = o[0][r] * inv;
    op[16] = o[1][r] * inv;
    op[32] = o[2][r] * inv;
    op[48] = o[3][r] * inv;
  }
}

// ---- launcher --------------------------------------------------------------

extern "C" void kernel_launch(void* const* d_in, const int* in_sizes, int n_in,
                              void* d_out, int out_size, void* d_ws, size_t ws_size,
                              hipStream_t stream) {
  const float* x  = (const float*)d_in[0];
  const float* Wq = (const float*)d_in[1];
  const float* Wk = (const float*)d_in[2];
  const float* Wv = (const float*)d_in[3];
  float* out = (float*)d_out;

  // workspace layout (bf16 = 2 bytes)
  char* ws = (char*)d_ws;
  unsigned short* xb  = (unsigned short*)(ws);                         // 4096*1024
  unsigned short* wqt = (unsigned short*)(ws + (size_t)8  * 1024 * 1024);
  unsigned short* wkt = (unsigned short*)(ws + (size_t)10 * 1024 * 1024);
  unsigned short* wvt = (unsigned short*)(ws + (size_t)12 * 1024 * 1024);
  unsigned short* Qb  = (unsigned short*)(ws + (size_t)14 * 1024 * 1024);
  unsigned short* Kb  = (unsigned short*)(ws + (size_t)22 * 1024 * 1024);
  unsigned short* Vt  = (unsigned short*)(ws + (size_t)30 * 1024 * 1024);

  const int nx = BN_TOK * DMODEL;        // 4,194,304
  cvt_bf16<<<2048, 256, 0, stream>>>(x, xb, nx);
  cvt_w_t <<<1024, 256, 0, stream>>>(Wq, wqt);
  cvt_w_t <<<1024, 256, 0, stream>>>(Wk, wkt);
  cvt_w_t <<<1024, 256, 0, stream>>>(Wv, wvt);

  gemm_qkv<<<dim3(BN_TOK / 128, DMODEL / 128, 3), 256, 0, stream>>>(
      xb, wqt, wkt, wvt, Qb, Kb, Vt);

  attn<<<dim3(SEQ / 128, BATCH * NHEAD), 256, 0, stream>>>(Qb, Kb, Vt, out);
}
